// TransformerBlock_70918499992241
// MI455X (gfx1250) — matrix operand order repaired and checked by reading, not compiled, not run on hardware
//
#include <hip/hip_runtime.h>
#include <hip/hip_bf16.h>

#define SEQ 2048
#define DIM 512
#define HEADS 8
#define DH 64
#define BATCH 4
#define MROWS (BATCH * SEQ)          // 8192
#define EPSF 1.1920929e-7f
#define INV_SQRT_DIM 0.044194173824159216f   // 512^-0.5

typedef _Float16 v16h __attribute__((ext_vector_type(16)));
typedef _Float16 v8h  __attribute__((ext_vector_type(8)));
typedef float    v8f  __attribute__((ext_vector_type(8)));

// ---------------------------------------------------------------- utilities
__global__ void __launch_bounds__(256) cvt_f16_kernel(const float* __restrict__ src,
                                                      _Float16* __restrict__ dst, int n) {
    int i = blockIdx.x * 256 + threadIdx.x;
    if (i < n) dst[i] = (_Float16)src[i];
}

// x += positional encoding; store f32 x; RMSNorm -> f16 h.  One block per row.
__global__ void __launch_bounds__(256) prep_kernel(const float* __restrict__ x,
                                                   const float* __restrict__ g,
                                                   const float* __restrict__ b,
                                                   float* __restrict__ xpos,
                                                   _Float16* __restrict__ h) {
    __shared__ float red[256];
    int row = blockIdx.x;              // 0..MROWS-1
    int s   = row & (SEQ - 1);
    int tid = threadIdx.x;
    int d0  = tid * 2;
    float e   = (float)d0 / (float)DIM;
    float div = 1.0f / (__powf(10000.0f, e) + EPSF);
    float ang = (float)s * div;
    size_t base = (size_t)row * DIM + d0;
    float v0 = x[base]     + __sinf(ang);
    float v1 = x[base + 1] + __cosf(ang);
    xpos[base]     = v0;
    xpos[base + 1] = v1;
    red[tid] = v0 * v0 + v1 * v1;
    __syncthreads();
    for (int st = 128; st > 0; st >>= 1) {
        if (tid < st) red[tid] += red[tid + st];
        __syncthreads();
    }
    float inv = 1.0f / (sqrtf(red[0]) * INV_SQRT_DIM + EPSF);
    h[base]     = (_Float16)(v0 * inv * g[d0]     + b[d0]);
    h[base + 1] = (_Float16)(v1 * inv * g[d0 + 1] + b[d0 + 1]);
}

// RMSNorm(f32 in) -> f16 out.  One block per row.
__global__ void __launch_bounds__(256) rmsnorm_kernel(const float* __restrict__ in,
                                                      const float* __restrict__ g,
                                                      const float* __restrict__ b,
                                                      _Float16* __restrict__ out) {
    __shared__ float red[256];
    int row = blockIdx.x;
    int tid = threadIdx.x;
    int d0  = tid * 2;
    size_t base = (size_t)row * DIM + d0;
    float v0 = in[base], v1 = in[base + 1];
    red[tid] = v0 * v0 + v1 * v1;
    __syncthreads();
    for (int st = 128; st > 0; st >>= 1) {
        if (tid < st) red[tid] += red[tid + st];
        __syncthreads();
    }
    float inv = 1.0f / (sqrtf(red[0]) * INV_SQRT_DIM + EPSF);
    out[base]     = (_Float16)(v0 * inv * g[d0]     + b[d0]);
    out[base + 1] = (_Float16)(v1 * inv * g[d0 + 1] + b[d0 + 1]);
}

// v[B,S,DIM] (f16) -> vT[B,H,DH,S] (f16)
__global__ void __launch_bounds__(256) vtrans_kernel(const _Float16* __restrict__ v,
                                                     _Float16* __restrict__ vT) {
    int idx = blockIdx.x * 256 + threadIdx.x;      // B*S*DIM = 2^22
    int c     = idx & (DIM - 1);
    int srow  = (idx >> 9) & (SEQ - 1);
    int batch = idx >> 20;
    int head  = c >> 6;
    int d     = c & (DH - 1);
    vT[(((size_t)(batch * HEADS + head)) * DH + d) * SEQ + srow] = v[idx];
}

// u[M,2*DIM] f32 -> out[M,DIM] f16:  silu(u[:,:D]) + u[:,D:]
__global__ void __launch_bounds__(256) swiglu_kernel(const float* __restrict__ u,
                                                     _Float16* __restrict__ out) {
    int idx = blockIdx.x * 256 + threadIdx.x;      // M*DIM
    int m = idx >> 9;
    int n = idx & (DIM - 1);
    float a = u[(size_t)m * (2 * DIM) + n];
    float c = u[(size_t)m * (2 * DIM) + DIM + n];
    float sig = 1.0f / (1.0f + __expf(-a));
    out[idx] = (_Float16)(a * sig + c);
}

// ------------------------------------------------------------ WMMA GEMM
// C[M,N] = A[M,K]f16 @ W[N,K]f16^T (+bias)(+residual) -> f16 or f32
// grid = (M/16, N/512), block = 256 (8 waves); wave w -> cols [w*64, w*64+64)
template<bool OUT_F16, bool HAS_BIAS, bool HAS_RES>
__global__ void __launch_bounds__(256) gemm_kernel(const _Float16* __restrict__ A,
                                                   const _Float16* __restrict__ W,
                                                   const float* __restrict__ bias,
                                                   const float* __restrict__ res,
                                                   _Float16* __restrict__ outH,
                                                   float* __restrict__ outF,
                                                   int K, int N) {
    int lane = threadIdx.x & 31;
    int wave = threadIdx.x >> 5;
    int r0 = blockIdx.x * 16;
    int c0 = blockIdx.y * 512 + wave * 64;
    int m16 = lane & 15;
    int hi  = (lane >> 4) & 1;
    v8f acc[4] = {};
    const _Float16* arow = A + (size_t)(r0 + m16) * K + hi * 16;
    for (int kk = 0; kk < K; kk += 32) {
        v16h af;
        {
            const _Float16* p = arow + kk;
            v8h lo = *(const v8h*)p;
            v8h hb = *(const v8h*)(p + 8);
#pragma unroll
            for (int i = 0; i < 8; i++) { af[i] = lo[i]; af[8 + i] = hb[i]; }
        }
#pragma unroll
        for (int nt = 0; nt < 4; nt++) {
            int n = c0 + nt * 16 + m16;
            v16h bf = *(const v16h*)(W + (size_t)n * K + kk + hi * 16);
            acc[nt] = __builtin_amdgcn_wmma_f32_16x16x32_f16(
                false, af, false, bf, (short)0, acc[nt], false, false);
        }
    }
#pragma unroll
    for (int nt = 0; nt < 4; nt++) {
        int n = c0 + nt * 16 + m16;
        float bv = HAS_BIAS ? bias[n] : 0.0f;
#pragma unroll
        for (int i = 0; i < 8; i++) {
            int m = r0 + i + hi * 8;
            float v = acc[nt][i] + bv;
            if (HAS_RES) v += res[(size_t)m * N + n];
            if (OUT_F16) outH[(size_t)m * N + n] = (_Float16)v;
            else         outF[(size_t)m * N + n] = v;
        }
    }
}

// ------------------------------------------------------- flash attention
// One wave = one 16-query tile of one head; 64 keys per iteration.
// Row sums computed by an extra "ones" WMMA tile (rides the O accumulate path);
// only the row max needs lane shuffles.  grid = B*H*(S/16)/8 blocks.
__global__ void __launch_bounds__(256) attn_kernel(const _Float16* __restrict__ Q,
                                                   const _Float16* __restrict__ Kmat,
                                                   const _Float16* __restrict__ Vt,
                                                   _Float16* __restrict__ O) {
    __shared__ __align__(32) _Float16 ldsP[8][16 * 64];   // 2 KB per wave
    int lane = threadIdx.x & 31;
    int wave = threadIdx.x >> 5;
    int tile = blockIdx.x * 8 + wave;                 // 0..4095
    int qt    = tile & 127;
    int head  = (tile >> 7) & 7;
    int batch = tile >> 10;
    int q0 = qt * 16;
    const _Float16* qh = Q    + (size_t)batch * SEQ * DIM + head * DH;
    const _Float16* kh = Kmat + (size_t)batch * SEQ * DIM + head * DH;
    const _Float16* vh = Vt   + ((size_t)(batch * HEADS + head)) * DH * SEQ;
    int m16 = lane & 15;
    int hi  = (lane >> 4) & 1;

    // Q A-fragments (16x64 = two 16x32 chunks), pre-scaled by DIM^-0.5
    const _Float16 qscale = (_Float16)INV_SQRT_DIM;
    v16h qa[2];
#pragma unroll
    for (int t = 0; t < 2; t++) {
        const _Float16* p = qh + (size_t)(q0 + m16) * DIM + t * 32 + hi * 16;
        v8h lo = *(const v8h*)p;
        v8h hb = *(const v8h*)(p + 8);
#pragma unroll
        for (int i = 0; i < 8; i++) {
            qa[t][i]     = lo[i] * qscale;
            qa[t][8 + i] = hb[i] * qscale;
        }
    }
    v16h ones;
#pragma unroll
    for (int i = 0; i < 16; i++) ones[i] = (_Float16)1.0f;

    v8f oacc[4] = {};
    v8f lacc = {};                                    // row sums (D-layout)
    float mrow[8];
#pragma unroll
    for (int i = 0; i < 8; i++) mrow[i] = -3.0e38f;

    int jEnd = q0 / 64 + 1;                           // keys [0, 64*jEnd)
    for (int j = 0; j < jEnd; j++) {
        int kb0 = j * 64;
        // ---- scores: four 16x16 tiles, each = 2 WMMAs over K=64
        v8f sacc[4] = {};
#pragma unroll
        for (int st = 0; st < 4; st++) {
            int key = kb0 + st * 16 + m16;
#pragma unroll
            for (int t = 0; t < 2; t++) {
                v16h bf = *(const v16h*)(kh + (size_t)key * DIM + t * 32 + hi * 16);
                sacc[st] = __builtin_amdgcn_wmma_f32_16x16x32_f16(
                    false, qa[t], false, bf, (short)0, sacc[st], false, false);
            }
        }
        // ---- causal mask + online-softmax rescale (rows = half-wave groups)
#pragma unroll
        for (int i = 0; i < 8; i++) {
            int qrow = q0 + i + hi * 8;
#pragma unroll
            for (int st = 0; st < 4; st++)
                if (kb0 + st * 16 + m16 > qrow) sacc[st][i] = -1.0e30f;
            float v = fmaxf(fmaxf(sacc[0][i], sacc[1][i]),
                            fmaxf(sacc[2][i], sacc[3][i]));
            for (int msk = 1; msk < 16; msk <<= 1) v = fmaxf(v, __shfl_xor(v, msk, 32));
            float mnew  = fmaxf(mrow[i], v);
            float alpha = __expf(mrow[i] - mnew);
            mrow[i] = mnew;
#pragma unroll
            for (int st = 0; st < 4; st++) sacc[st][i] = __expf(sacc[st][i] - mnew);
#pragma unroll
            for (int nt = 0; nt < 4; nt++) oacc[nt][i] *= alpha;
            lacc[i] *= alpha;
        }
        // ---- P: D-layout -> LDS -> two A-fragments (per-wave private patch)
        _Float16* lp = ldsP[wave];
#pragma unroll
        for (int st = 0; st < 4; st++)
#pragma unroll
            for (int i = 0; i < 8; i++)
                lp[(i + hi * 8) * 64 + st * 16 + m16] = (_Float16)sacc[st][i];
        asm volatile("s_wait_dscnt 0" ::: "memory");
        v16h pa[2];
#pragma unroll
        for (int t2 = 0; t2 < 2; t2++) {
            const _Float16* rp = lp + m16 * 64 + t2 * 32 + hi * 16;
            v8h lo = *(const v8h*)rp;
            v8h hb = *(const v8h*)(rp + 8);
#pragma unroll
            for (int i = 0; i < 8; i++) { pa[t2][i] = lo[i]; pa[t2][8 + i] = hb[i]; }
        }
        asm volatile("s_wait_dscnt 0" ::: "memory");
        // ---- O += P @ V ; l += P @ 1   (V pre-transposed: contiguous keys)
#pragma unroll
        for (int t2 = 0; t2 < 2; t2++) {
            lacc = __builtin_amdgcn_wmma_f32_16x16x32_f16(
                false, pa[t2], false, ones, (short)0, lacc, false, false);
#pragma unroll
            for (int nt = 0; nt < 4; nt++) {
                v16h vb = *(const v16h*)(vh + (size_t)(nt * 16 + m16) * SEQ
                                            + kb0 + t2 * 32 + hi * 16);
                oacc[nt] = __builtin_amdgcn_wmma_f32_16x16x32_f16(
                    false, pa[t2], false, vb, (short)0, oacc[nt], false, false);
            }
        }
    }
    // ---- epilogue: O /= l, store f16
#pragma unroll
    for (int nt = 0; nt < 4; nt++)
#pragma unroll
        for (int i = 0; i < 8; i++) {
            int m = q0 + i + hi * 8;
            float val = oacc[nt][i] / lacc[i];
            O[(size_t)(batch * SEQ + m) * DIM + head * DH + nt * 16 + m16] = (_Float16)val;
        }
}

// ------------------------------------------------------------- launcher
extern "C" void kernel_launch(void* const* d_in, const int* in_sizes, int n_in,
                              void* d_out, int out_size, void* d_ws, size_t ws_size,
                              hipStream_t stream) {
    const float* x       = (const float*)d_in[0];
    const float* norm1_g = (const float*)d_in[1];
    const float* norm1_b = (const float*)d_in[2];
    const float* wq      = (const float*)d_in[3];
    const float* wk      = (const float*)d_in[4];
    const float* wv      = (const float*)d_in[5];
    const float* wo      = (const float*)d_in[6];
    const float* norm2_g = (const float*)d_in[7];
    const float* norm2_b = (const float*)d_in[8];
    const float* ff_w1   = (const float*)d_in[9];
    const float* ff_b1   = (const float*)d_in[10];
    const float* ffn_g   = (const float*)d_in[11];
    const float* ffn_b   = (const float*)d_in[12];
    const float* sw_w    = (const float*)d_in[13];
    const float* sw_b    = (const float*)d_in[14];
    const float* ff_w2   = (const float*)d_in[15];
    const float* ff_b2   = (const float*)d_in[16];

    char* ws = (char*)d_ws;
    size_t off = 0;
    auto alloc = [&](size_t bytes) -> char* {
        off = (off + 255) & ~(size_t)255;
        char* p = ws + off;
        off += bytes;
        return p;
    };
    const size_t SZ_W  = (size_t)DIM * DIM * 2;          // f16 512x512
    const size_t SZ_H  = (size_t)MROWS * DIM * 2;        // f16 activations
    const size_t SZ_F  = (size_t)MROWS * DIM * 4;        // f32 activations

    _Float16* wq16 = (_Float16*)alloc(SZ_W);
    _Float16* wk16 = (_Float16*)alloc(SZ_W);
    _Float16* wv16 = (_Float16*)alloc(SZ_W);
    _Float16* wo16 = (_Float16*)alloc(SZ_W);
    _Float16* w116 = (_Float16*)alloc(SZ_W);
    _Float16* w216 = (_Float16*)alloc(SZ_W);
    _Float16* sw16 = (_Float16*)alloc(2 * SZ_W);
    float*    xpos = (float*)alloc(SZ_F);                // later aliased by ff1
    _Float16* hbuf = (_Float16*)alloc(SZ_H);             // reused 3x
    char*     qreg = alloc(4 * SZ_H);                    // q,k,v,vT; later u f32
    _Float16* q16  = (_Float16*)qreg;
    _Float16* k16  = (_Float16*)(qreg + SZ_H);
    _Float16* v16  = (_Float16*)(qreg + 2 * SZ_H);
    _Float16* vT16 = (_Float16*)(qreg + 3 * SZ_H);
    float*    ubuf = (float*)qreg;                       // [M, 2*DIM] f32
    _Float16* obuf = (_Float16*)alloc(SZ_H);             // attn out; later h4
    _Float16* h4   = obuf;
    float*    x2   = (float*)alloc(SZ_F);
    float*    ff1  = xpos;                               // alias (xpos dead after wo)

    // 1) weights -> f16
    int nw = DIM * DIM;
    cvt_f16_kernel<<<(nw + 255) / 256, 256, 0, stream>>>(wq, wq16, nw);
    cvt_f16_kernel<<<(nw + 255) / 256, 256, 0, stream>>>(wk, wk16, nw);
    cvt_f16_kernel<<<(nw + 255) / 256, 256, 0, stream>>>(wv, wv16, nw);
    cvt_f16_kernel<<<(nw + 255) / 256, 256, 0, stream>>>(wo, wo16, nw);
    cvt_f16_kernel<<<(nw + 255) / 256, 256, 0, stream>>>(ff_w1, w116, nw);
    cvt_f16_kernel<<<(nw + 255) / 256, 256, 0, stream>>>(ff_w2, w216, nw);
    cvt_f16_kernel<<<(2 * nw + 255) / 256, 256, 0, stream>>>(sw_w, sw16, 2 * nw);

    // 2) x += pos; h = rmsnorm1(x)
    prep_kernel<<<MROWS, 256, 0, stream>>>(x, norm1_g, norm1_b, xpos, hbuf);

    // 3) Q/K/V projections (f16 out)
    dim3 g1(MROWS / 16, 1);
    gemm_kernel<true, false, false><<<g1, 256, 0, stream>>>(hbuf, wq16, nullptr, nullptr, q16, nullptr, DIM, DIM);
    gemm_kernel<true, false, false><<<g1, 256, 0, stream>>>(hbuf, wk16, nullptr, nullptr, k16, nullptr, DIM, DIM);
    gemm_kernel<true, false, false><<<g1, 256, 0, stream>>>(hbuf, wv16, nullptr, nullptr, v16, nullptr, DIM, DIM);

    // 4) V -> [B,H,DH,S]
    vtrans_kernel<<<(MROWS * DIM) / 256, 256, 0, stream>>>(v16, vT16);

    // 5) flash attention -> obuf f16
    attn_kernel<<<(BATCH * HEADS * (SEQ / 16)) / 8, 256, 0, stream>>>(q16, k16, vT16, obuf);

    // 6) x2 = obuf @ wo^T + xpos
    gemm_kernel<false, false, true><<<g1, 256, 0, stream>>>(obuf, wo16, nullptr, xpos, nullptr, x2, DIM, DIM);

    // 7) hbuf = rmsnorm2(x2)
    rmsnorm_kernel<<<MROWS, 256, 0, stream>>>(x2, norm2_g, norm2_b, hbuf);

    // 8) ff1 = hbuf @ ff_w1^T + b1 (f32)
    gemm_kernel<false, true, false><<<g1, 256, 0, stream>>>(hbuf, w116, ff_b1, nullptr, nullptr, ff1, DIM, DIM);

    // 9) hbuf = rmsnorm_ffn(ff1)
    rmsnorm_kernel<<<MROWS, 256, 0, stream>>>(ff1, ffn_g, ffn_b, hbuf);

    // 10) u = hbuf @ sw_w^T + sw_b  (N=1024, f32)
    dim3 g2(MROWS / 16, 2);
    gemm_kernel<false, true, false><<<g2, 256, 0, stream>>>(hbuf, sw16, sw_b, nullptr, nullptr, ubuf, DIM, 2 * DIM);

    // 11) h4 = silu(u[:, :512]) + u[:, 512:]
    swiglu_kernel<<<(MROWS * DIM) / 256, 256, 0, stream>>>(ubuf, h4);

    // 12) out = h4 @ ff_w2^T + b2 + x2
    gemm_kernel<false, true, true><<<g1, 256, 0, stream>>>(h4, w216, ff_b2, x2, nullptr, (float*)d_out, DIM, DIM);

    (void)in_sizes; (void)n_in; (void)out_size; (void)ws_size;
}
